// PIC_78194174591154
// MI455X (gfx1250) — matrix operand order repaired and checked by reading, not compiled, not run on hardware
//
#include <hip/hip_runtime.h>
#include <hip/hip_bf16.h>
#include <math.h>

typedef __attribute__((ext_vector_type(16))) _Float16 v16h;
typedef __attribute__((ext_vector_type(8)))  _Float16 v8h;
typedef __attribute__((ext_vector_type(8)))  float    v8f;

#define NIP_     256
#define N_INNER_ 16
#define NET_DIM_ 64
#define N_INPUT_ 784
#define N_CAT_   256
#define TWO_PI_  6.283185307179586f
#define LOG2E2_  2.885390081777927f   // 2*log2(e)

union AOp { v16h v; _Float16 h[16]; };
union HOp { v8h  v; _Float16 h[8];  };

// branchless tanh: 1 - 2/(exp(2x)+1); exp2 overflow -> inf -> rcp -> 0 -> 1
__device__ __forceinline__ float tanh_fast(float x) {
  const float e = __builtin_amdgcn_exp2f(x * LOG2E2_);
  return 1.0f - 2.0f * __builtin_amdgcn_rcpf(e + 1.0f);
}

// stable softplus with fast natives: max(x,0) + log(1 + exp(-|x|))
__device__ __forceinline__ float softplus_f(float x) {
  return fmaxf(x, 0.0f) + __logf(1.0f + __expf(-fabsf(x)));
}

// Load 8 consecutive f32 from a 32B-aligned row pointer, convert to f16.
__device__ __forceinline__ void load_a8(const float* __restrict__ rowp, int k0,
                                        _Float16* dst) {
  const float4 f0 = *(const float4*)(rowp + k0);
  const float4 f1 = *(const float4*)(rowp + k0 + 4);
  dst[0] = (_Float16)f0.x; dst[1] = (_Float16)f0.y;
  dst[2] = (_Float16)f0.z; dst[3] = (_Float16)f0.w;
  dst[4] = (_Float16)f1.x; dst[5] = (_Float16)f1.y;
  dst[6] = (_Float16)f1.z; dst[7] = (_Float16)f1.w;
}

__device__ __forceinline__ v8f wmma_f16(v16h a, v16h b, v8f c) {
  return __builtin_amdgcn_wmma_f32_16x16x32_f16(false, a, false, b,
                                                (short)0, c, false, false);
}

// ---------------------------------------------------------------------------
// Kernel A: inner net.  One block per i (pair row).  256 threads = 8 waves.
// ---------------------------------------------------------------------------
__global__ __launch_bounds__(256)
void PIC_inner_kernel(const float* __restrict__ z,
                      const float* __restrict__ log_w,
                      const float* __restrict__ coeff,   // [2,32]
                      const float* __restrict__ w1,      // [1024,64]
                      const float* __restrict__ w2,      // [1024]
                      float* __restrict__ out)           // [16,256,256]
{
  __shared__ _Float16 ffT[NIP_ * NET_DIM_];   // [j][k], f16, 32KB
  __shared__ float    acc[N_INNER_ * NIP_];   // [g][j], f32, 16KB

  const int i    = blockIdx.x;
  const int tid  = threadIdx.x;
  const int lane = tid & 31;
  const int wv   = tid >> 5;
  const float zi = z[i];

  // ---- phase 1: fourier features for this i, all j; zero accumulators ----
  {
    const float zj = z[tid];
    #pragma unroll
    for (int kk = 0; kk < 32; ++kk) {
      const float xp = TWO_PI_ * (zi * coeff[kk] + zj * coeff[32 + kk]);
      float s, c;
      __sincosf(xp, &s, &c);
      ffT[tid * 64 + kk]      = (_Float16)c;
      ffT[tid * 64 + 32 + kk] = (_Float16)s;
    }
    #pragma unroll
    for (int q = 0; q < 16; ++q) acc[q * 256 + tid] = 0.0f;
  }
  __syncthreads();

  // ---- phase 2: 1024x256x64 GEMM + tanh + w2 contraction ----
  const int  mrow    = lane & 15;
  const bool halfsel = (lane >= 16);
  const int  aoff    = halfsel ? 8 : 0;
  const int  kb0     = halfsel ? 16 : 0;

  float p[2][16];
  #pragma unroll
  for (int gl = 0; gl < 2; ++gl)
    #pragma unroll
    for (int t = 0; t < 16; ++t) p[gl][t] = 0.0f;

  #pragma unroll
  for (int gl = 0; gl < 2; ++gl) {       // unrolled: p[gl] register-indexed
    #pragma unroll 1
    for (int mm4 = 0; mm4 < 4; ++mm4) {  // rolled: A/B/w2 regs reused
      const int m = wv * 8 + gl * 4 + mm4;   // M-tile 0..63
      const float* rp = w1 + (size_t)(m * 16 + mrow) * 64;
      AOp a0, a1;
      load_a8(rp,  0 + kb0, a0.h);
      load_a8(rp,  8 + kb0, a0.h + 8);
      load_a8(rp, 32 + kb0, a1.h);
      load_a8(rp, 40 + kb0, a1.h + 8);

      // w2 entries for this lane's 8 D rows
      const int drow0 = m * 16 + aoff;
      float w2r[8];
      {
        const float4 f0 = *(const float4*)(w2 + drow0);
        const float4 f1 = *(const float4*)(w2 + drow0 + 4);
        w2r[0] = f0.x; w2r[1] = f0.y; w2r[2] = f0.z; w2r[3] = f0.w;
        w2r[4] = f1.x; w2r[5] = f1.y; w2r[6] = f1.z; w2r[7] = f1.w;
      }

      #pragma unroll
      for (int t = 0; t < 16; ++t) {
        const int j = t * 16 + mrow;
        const v16h b0 = *(const v16h*)&ffT[j * 64 + kb0];
        const v16h b1 = *(const v16h*)&ffT[j * 64 + 32 + kb0];
        v8f c = {0.f, 0.f, 0.f, 0.f, 0.f, 0.f, 0.f, 0.f};
        c = wmma_f16(a0.v, b0, c);
        c = wmma_f16(a1.v, b1, c);
        float s = 0.0f;
        #pragma unroll
        for (int r = 0; r < 8; ++r) s += w2r[r] * tanh_fast(c[r]);
        p[gl][t] += s;
      }
    }
  }

  // reduce partials into LDS accumulator (lane pairs + 4 m-tiles per g)
  const int g0 = wv * 2;
  #pragma unroll
  for (int gl = 0; gl < 2; ++gl)
    #pragma unroll
    for (int t = 0; t < 16; ++t)
      atomicAdd(&acc[(g0 + gl) * 256 + t * 16 + mrow], p[gl][t]);
  __syncthreads();

  // ---- phase 3: softplus, logsumexp over j, write transposed output ----
  #pragma unroll
  for (int gl = 0; gl < 2; ++gl) {
    const int g = wv * 2 + gl;
    float tv[8];
    float mx = -INFINITY;
    #pragma unroll
    for (int q = 0; q < 8; ++q) {
      const int j = lane * 8 + q;
      const float lv = -softplus_f(acc[g * 256 + j]);
      tv[q] = lv + log_w[j];
      mx = fmaxf(mx, tv[q]);
    }
    #pragma unroll
    for (int off = 16; off > 0; off >>= 1)
      mx = fmaxf(mx, __shfl_xor(mx, off, 32));
    float se = 0.0f;
    #pragma unroll
    for (int q = 0; q < 8; ++q) se += __expf(tv[q] - mx);
    #pragma unroll
    for (int off = 16; off > 0; off >>= 1)
      se += __shfl_xor(se, off, 32);
    const float lse = mx + __logf(se);
    #pragma unroll
    for (int q = 0; q < 8; ++q) {
      const int j = lane * 8 + q;
      // inner_param[g, j, i] = exp(logits - lse + log_w[j]) = exp(tv - lse)
      out[((size_t)g * 256 + j) * 256 + i] = __expf(tv[q] - lse);
    }
  }
}

// ---------------------------------------------------------------------------
// Kernel C: input net.  One block per g (0..783).  256 threads = 8 waves.
// Dynamic LDS 160KB: logitsT[n][c] f16 (128KB, aliases ffi region) +
// hiT[n][d] f16 (32KB).
// ---------------------------------------------------------------------------
__global__ __launch_bounds__(256)
void PIC_input_kernel(const float* __restrict__ z,
                      const float* __restrict__ icoeff,  // [1,32]
                      const float* __restrict__ w1,      // [784*64, 64]
                      const float* __restrict__ w2,      // [784,256,64]
                      float* __restrict__ out)           // [784,256,256]
{
  extern __shared__ _Float16 smem[];
  _Float16* logitsT = smem;                    // [n][c]  65536 halves
  _Float16* ffT     = smem;                    // [n][k]  (aliases logitsT)
  _Float16* hiT     = smem + 65536;            // [n][d]  16384 halves

  const int g    = blockIdx.x;
  const int tid  = threadIdx.x;
  const int lane = tid & 31;
  const int wv   = tid >> 5;

  // ---- phase 1: fourier features of z (input net) ----
  {
    const float zn = z[tid];
    #pragma unroll
    for (int kk = 0; kk < 32; ++kk) {
      const float xp = TWO_PI_ * zn * icoeff[kk];
      float s, c;
      __sincosf(xp, &s, &c);
      ffT[tid * 64 + kk]      = (_Float16)c;
      ffT[tid * 64 + 32 + kk] = (_Float16)s;
    }
  }
  __syncthreads();

  const int  nrow    = lane & 15;
  const bool halfsel = (lane >= 16);
  const int  aoff    = halfsel ? 8 : 0;
  const int  kb0     = halfsel ? 16 : 0;

  // ---- phase 2: hi = tanh(W1_g @ ffi), stored f16 as hiT[n][d] ----
  {
    const int mmh   = wv >> 1;            // d-tile 0..3
    const int tbase = (wv & 1) * 8;       // n-tiles 0..7 or 8..15
    const float* rp = w1 + ((size_t)g * 64 + mmh * 16 + nrow) * 64;
    AOp a0, a1;
    load_a8(rp,  0 + kb0, a0.h);
    load_a8(rp,  8 + kb0, a0.h + 8);
    load_a8(rp, 32 + kb0, a1.h);
    load_a8(rp, 40 + kb0, a1.h + 8);
    const int d0 = mmh * 16 + aoff;
    #pragma unroll 1
    for (int tt = 0; tt < 8; ++tt) {
      const int n = (tbase + tt) * 16 + nrow;
      const v16h b0 = *(const v16h*)&ffT[n * 64 + kb0];
      const v16h b1 = *(const v16h*)&ffT[n * 64 + 32 + kb0];
      v8f c = {0.f, 0.f, 0.f, 0.f, 0.f, 0.f, 0.f, 0.f};
      c = wmma_f16(a0.v, b0, c);
      c = wmma_f16(a1.v, b1, c);
      HOp hv;
      #pragma unroll
      for (int r = 0; r < 8; ++r) hv.h[r] = (_Float16)tanh_fast(c[r]);
      *(v8h*)&hiT[n * 64 + d0] = hv.v;
    }
  }
  __syncthreads();

  // ---- phase 3: in_logits = W2_g @ hi, stored f16 as logitsT[n][c] ----
  {
    AOp a[2][2];
    #pragma unroll
    for (int mi = 0; mi < 2; ++mi) {
      const int mm = wv * 2 + mi;          // c-tile 0..15
      const float* rp = w2 + (((size_t)g * 256) + mm * 16 + nrow) * 64;
      load_a8(rp,  0 + kb0, a[mi][0].h);
      load_a8(rp,  8 + kb0, a[mi][0].h + 8);
      load_a8(rp, 32 + kb0, a[mi][1].h);
      load_a8(rp, 40 + kb0, a[mi][1].h + 8);
    }
    #pragma unroll 1
    for (int t = 0; t < 16; ++t) {
      const int n = t * 16 + nrow;
      const v16h b0 = *(const v16h*)&hiT[n * 64 + kb0];
      const v16h b1 = *(const v16h*)&hiT[n * 64 + 32 + kb0];
      #pragma unroll
      for (int mi = 0; mi < 2; ++mi) {
        v8f c = {0.f, 0.f, 0.f, 0.f, 0.f, 0.f, 0.f, 0.f};
        c = wmma_f16(a[mi][0].v, b0, c);
        c = wmma_f16(a[mi][1].v, b1, c);
        const int c0 = (wv * 2 + mi) * 16 + aoff;
        HOp hv;
        #pragma unroll
        for (int r = 0; r < 8; ++r) hv.h[r] = (_Float16)c[r];
        *(v8h*)&logitsT[n * 256 + c0] = hv.v;
      }
    }
  }
  __syncthreads();

  // ---- phase 4: log_softmax over c, coalesced stores ----
  {
    float* outg = out + (size_t)g * 256 * 256;
    #pragma unroll 1
    for (int it = 0; it < 32; ++it) {
      const int n = it * 8 + wv;
      HOp hv;
      hv.v = *(const v8h*)&logitsT[n * 256 + lane * 8];
      float v[8];
      float mx = -INFINITY;
      #pragma unroll
      for (int q = 0; q < 8; ++q) { v[q] = (float)hv.h[q]; mx = fmaxf(mx, v[q]); }
      #pragma unroll
      for (int off = 16; off > 0; off >>= 1)
        mx = fmaxf(mx, __shfl_xor(mx, off, 32));
      float se = 0.0f;
      #pragma unroll
      for (int q = 0; q < 8; ++q) se += __expf(v[q] - mx);
      #pragma unroll
      for (int off = 16; off > 0; off >>= 1)
        se += __shfl_xor(se, off, 32);
      const float lse = mx + __logf(se);
      float4 o0 = {v[0] - lse, v[1] - lse, v[2] - lse, v[3] - lse};
      float4 o1 = {v[4] - lse, v[5] - lse, v[6] - lse, v[7] - lse};
      *(float4*)&outg[(size_t)n * 256 + lane * 8]     = o0;
      *(float4*)&outg[(size_t)n * 256 + lane * 8 + 4] = o1;
    }
  }
}

extern "C" void kernel_launch(void* const* d_in, const int* in_sizes, int n_in,
                              void* d_out, int out_size, void* d_ws, size_t ws_size,
                              hipStream_t stream) {
  (void)in_sizes; (void)n_in; (void)out_size; (void)d_ws; (void)ws_size;
  const float* z           = (const float*)d_in[0];
  const float* log_w       = (const float*)d_in[1];
  const float* inner_coeff = (const float*)d_in[2];
  const float* inner_w1    = (const float*)d_in[3];
  const float* inner_w2    = (const float*)d_in[4];
  const float* input_coeff = (const float*)d_in[5];
  const float* input_w1    = (const float*)d_in[6];
  const float* input_w2    = (const float*)d_in[7];

  float* out_inner = (float*)d_out;                        // [16,256,256]
  float* out_input = out_inner + N_INNER_ * NIP_ * NIP_;   // [784,256,256]

  PIC_inner_kernel<<<NIP_, 256, 0, stream>>>(
      z, log_w, inner_coeff, inner_w1, inner_w2, out_inner);

  PIC_input_kernel<<<N_INPUT_, 256, 160 * 1024, stream>>>(
      z, input_coeff, input_w1, input_w2, out_input);
}
